// BiLSTM_CRF_31636729103069
// MI455X (gfx1250) — matrix operand order repaired and checked by reading, not compiled, not run on hardware
//
#include <hip/hip_runtime.h>
#include <hip/hip_bf16.h>

// ---------------------------------------------------------------------------
// BiLSTM-CRF for MI455X (gfx1250, wave32, WMMA + TDM).
//   B=128 T=512 E=300 H=50 G=4H=200 K=7
// Pipeline:
//   1) emb gather -> f16 x [BT,320]           (pad E 300->320)
//   2) per layer: pack weights f32->f16 padded; WMMA GEMM xg = x@w_ih^T + b
//      (M=65536, Kpad in {320,128}, Npad=208, both dirs) -> f16 [2][BT][208]
//      A-block staged into LDS by the Tensor Data Mover (tensor_load_to_lds).
//   3) per layer: persistent LSTM recurrence, 16 WGs = 2 dirs x 8 batch tiles,
//      h kept in LDS (16x64 f16), w_hh WMMA B-fragments resident in VGPRs,
//      2x v_wmma_f32_16x16x32_f16 per N-tile per timestep, f32 cell state.
//      xg timestep tiles (16x208 f16, strided) double-buffered in LDS via TDM
//      one step ahead (s_wait_tensorcnt<=1) to hide latency on the sequential
//      critical path.
//   4) Viterbi: one wave32 per batch row; K=7 lanes, bptr bytes in scratch.
// Workspace use: ~131 MB.
// ---------------------------------------------------------------------------

typedef __attribute__((ext_vector_type(16))) _Float16 v16h;
typedef __attribute__((ext_vector_type(8)))  _Float16 v8h;
typedef __attribute__((ext_vector_type(8)))  float    v8f;
typedef __attribute__((ext_vector_type(4)))  unsigned int v4u;
typedef __attribute__((ext_vector_type(8)))  int v8i;
typedef __attribute__((ext_vector_type(4)))  int v4i;

constexpr int Bsz  = 128;
constexpr int Tlen = 512;
constexpr int Esz  = 300;
constexpr int Hsz  = 50;
constexpr int Gsz  = 200;   // 4*H
constexpr int Npad = 208;   // G padded to 13 WMMA tiles
constexpr long BT  = (long)Bsz * Tlen;
constexpr float NEGF = -10000.0f;

// ---- WMMA fragment helpers (CDNA5 16x16x32 f16, wave32; ISA 7.12.2) -------
// A: 16x32 (MxK) f16, row-major source with leading dim ld (halves).
//    lane l: M = l&15, hi = l>>4; halves 0..7 -> K = hi*8 + 0..7,
//    halves 8..15 -> K = 16 + hi*8 + 0..7. Two contiguous 16B loads.
__device__ __forceinline__ v16h load_a_frag(const _Float16* p, int ld) {
  int l = threadIdx.x & 31;
  const _Float16* r = p + (l & 15) * ld + (l >> 4) * 8;
  v8h lo = *(const v8h*)(r);
  v8h hi = *(const v8h*)(r + 16);
  v16h a;
#pragma unroll
  for (int k = 0; k < 8; ++k) { a[k] = lo[k]; a[8 + k] = hi[k]; }
  return a;
}

// B: 32x16 (KxN) f16 built from weights stored [N,K] row-major (so B = W^T).
//    lane l: N = l&15, hi = l>>4; halves 0..7 -> K = hi*8 + 0..7,
//    halves 8..15 -> K = 16 + hi*8 + 0..7. Two contiguous 16B loads.
__device__ __forceinline__ v16h load_b_frag(const _Float16* p, int ld) {
  int l = threadIdx.x & 31;
  const _Float16* r = p + (l & 15) * ld + (l >> 4) * 8;
  v8h lo = *(const v8h*)(r);
  v8h hi = *(const v8h*)(r + 16);
  v16h b;
#pragma unroll
  for (int k = 0; k < 8; ++k) { b[k] = lo[k]; b[8 + k] = hi[k]; }
  return b;
}

__device__ __forceinline__ v8f wmma_f16(v16h a, v16h b, v8f c) {
  // v_wmma_f32_16x16x32_f16 (8-arg form: neg_a, A, neg_b, B, c_mod, C, reuse)
  return __builtin_amdgcn_wmma_f32_16x16x32_f16(false, a, false, b,
                                                (short)0, c, false, false);
}

__device__ __forceinline__ float sigmoidf(float x) {
  return 1.0f / (1.0f + __expf(-x));
}

// ---- Tensor Data Mover: 2-D f16 tile (tile_rows x tile_cols), row stride in
// elements, global -> LDS. Descriptor per CDNA5 ISA ch.8 (groups 0/1; 2-D so
// groups 2/3 are zero). Issued once per wave; tracked by TENSORcnt.
#if defined(__AMDGCN__) && __has_builtin(__builtin_amdgcn_tensor_load_to_lds)
#define HAVE_TDM 1
__device__ __forceinline__ unsigned lds_off_of(const void* p) {
  // generic "shared" pointers carry the LDS byte offset in the low 32 bits
  return (unsigned)(unsigned long)p;
}
__device__ __forceinline__ void tdm_load_2d(unsigned lds_off, const void* gptr,
                                            unsigned tile_cols, unsigned tile_rows,
                                            unsigned long row_stride_elems) {
  unsigned long ga = (unsigned long)gptr;
  v4u g0;
  g0[0] = 1u;                                   // count=1 (valid), no gather
  g0[1] = lds_off;                              // lds_addr (bytes)
  g0[2] = (unsigned)ga;                         // global_addr[31:0]
  g0[3] = (unsigned)((ga >> 32) & 0x01FFFFFFu)  // global_addr[56:32]
        | (2u << 30);                           // type=2 ("image")
  v8i g1;
  unsigned td0 = tile_cols, td1 = tile_rows;    // tensor dims == tile (no OOB)
  g1[0] = (int)(1u << 16);                      // wg_mask=0, data_size=1 (2B)
  g1[1] = (int)((td0 & 0xFFFFu) << 16);         // tensor_dim0[15:0]
  g1[2] = (int)((td0 >> 16) | ((td1 & 0xFFFFu) << 16));  // dim0 hi | dim1 lo
  g1[3] = (int)((td1 >> 16) | ((tile_cols & 0xFFFFu) << 16)); // tile_dim0
  g1[4] = (int)(tile_rows & 0xFFFFu);           // tile_dim1; tile_dim2=0
  g1[5] = (int)(unsigned)(row_stride_elems & 0xFFFFFFFFu);    // dim0 stride lo
  g1[6] = (int)((row_stride_elems >> 32) & 0xFFFFu);          // stride hi
  g1[7] = 0;
  v4i z = {0, 0, 0, 0};
#if __clang_major__ >= 23
  v8i z8 = {0, 0, 0, 0, 0, 0, 0, 0};
  __builtin_amdgcn_tensor_load_to_lds(g0, g1, z, z, z8, 0);
#else
  __builtin_amdgcn_tensor_load_to_lds(g0, g1, z, z, 0);
#endif
}
#endif

// ---- 1) Embedding gather: x[b*T+t][0..319] = emb[sentence[b,t]] (f16) -----
__global__ void __launch_bounds__(256)
emb_gather(const int* __restrict__ sent, const float* __restrict__ emb,
           _Float16* __restrict__ X) {
  long row = (long)blockIdx.x * 8 + (threadIdx.x >> 5);
  int lane = threadIdx.x & 31;
  int v = sent[row];
  const float* src = emb + (long)v * Esz;
  _Float16* dst = X + row * 320;
  for (int e = lane; e < 320; e += 32)
    dst[e] = (e < Esz) ? (_Float16)src[e] : (_Float16)0.0f;
}

// ---- weight repack: f32 -> f16, zero-padded [2][208][Kpad] / [2][208][64] -
__global__ void pack_weights(const float* __restrict__ wih,
                             const float* __restrict__ whh,
                             const float* __restrict__ bias,
                             _Float16* __restrict__ WIH,
                             _Float16* __restrict__ WHH,
                             float* __restrict__ BIA,
                             int Kin, int Kpad) {
  int nIH = 2 * Npad * Kpad;
  int nHH = 2 * Npad * 64;
  int total = nIH + nHH + 2 * Npad;
  for (int i = blockIdx.x * blockDim.x + threadIdx.x; i < total;
       i += gridDim.x * blockDim.x) {
    if (i < nIH) {
      int d = i / (Npad * Kpad), rem = i % (Npad * Kpad);
      int n = rem / Kpad, k = rem % Kpad;
      float v = (n < Gsz && k < Kin) ? wih[((long)d * Gsz + n) * Kin + k] : 0.0f;
      WIH[i] = (_Float16)v;
    } else if (i < nIH + nHH) {
      int j = i - nIH;
      int d = j / (Npad * 64), rem = j % (Npad * 64);
      int n = rem / 64, k = rem % 64;
      float v = (n < Gsz && k < Hsz) ? whh[((long)d * Gsz + n) * Hsz + k] : 0.0f;
      WHH[j] = (_Float16)v;
    } else {
      int j = i - nIH - nHH;
      int d = j / Npad, n = j % Npad;
      BIA[j] = (n < Gsz) ? bias[d * Gsz + n] : 0.0f;
    }
  }
}

// ---- 2) Input projection GEMM: xg = x @ w_ih^T + b (both dirs) ------------
// grid: BT/128 blocks, 256 threads (8 waves). A block (128 x Kpad) staged in
// LDS by TDM; each wave owns one 16-row M-tile and loops 26 N-tiles x K.
__global__ void __launch_bounds__(256)
gemm_xg(const _Float16* __restrict__ X, const _Float16* __restrict__ W,
        const float* __restrict__ BIA, _Float16* __restrict__ XG, int Kpad) {
  extern __shared__ _Float16 sA[];  // [128][Kpad]
  int tid = threadIdx.x;
  long m0 = (long)blockIdx.x * 128;
#ifdef HAVE_TDM
  if (tid < 32) {  // wave 0 drives the DMA; transfer is contiguous 128*Kpad
    tdm_load_2d(lds_off_of(sA), X + m0 * Kpad, (unsigned)Kpad, 128u,
                (unsigned long)Kpad);
    __builtin_amdgcn_s_wait_tensorcnt(0);
  }
#else
  int elems = 128 * Kpad;
  for (int i = tid * 8; i < elems; i += 256 * 8)
    *(v8h*)(sA + i) = *(const v8h*)(X + m0 * Kpad + i);
#endif
  __syncthreads();

  int wave = tid >> 5, lane = tid & 31;
  int mrow = wave * 16;
  int nloc = lane & 15, hi2 = lane >> 4;
  for (int nt = 0; nt < 26; ++nt) {
    int dir = nt / 13, n0 = (nt % 13) * 16;
    float bb = BIA[dir * Npad + n0 + nloc];
    v8f acc;
#pragma unroll
    for (int r = 0; r < 8; ++r) acc[r] = bb;  // bias pre-folded into C
    const _Float16* wbase = W + ((long)dir * Npad + n0) * Kpad;
    for (int kb = 0; kb < Kpad; kb += 32) {
      __builtin_prefetch(wbase + (long)nloc * Kpad + kb + 64, 0, 1);
      v16h a = load_a_frag(sA + mrow * Kpad + kb, Kpad);
      v16h b = load_b_frag(wbase + kb, Kpad);
      acc = wmma_f16(a, b, acc);
    }
    _Float16* out = XG + ((long)dir * BT + m0 + mrow) * Npad;
#pragma unroll
    for (int r = 0; r < 8; ++r)
      out[(long)(r + hi2 * 8) * Npad + n0 + nloc] = (_Float16)acc[r];
  }
}

// ---- 3) Persistent LSTM recurrence ----------------------------------------
// grid 16: blockIdx = dir + 2*btile. 416 threads = 13 waves; wave nt owns
// gate N-tile nt. h (16x64 f16) lives in LDS; w_hh fragments live in VGPRs.
// xg timestep tiles are TDM-prefetched one step ahead into a double buffer.
__global__ void __launch_bounds__(416)
lstm_layer(const _Float16* __restrict__ XG, const _Float16* __restrict__ WHH,
           const float* __restrict__ h0, const float* __restrict__ c0,
           _Float16* __restrict__ HC) {
  __shared__ __align__(32) _Float16 hA[16 * 64];   // h state, A operand
  __shared__ float gbuf[16 * Npad];                // gate preactivations
  __shared__ float cbuf[16 * 52];                  // cell state
#ifdef HAVE_TDM
  __shared__ __align__(32) _Float16 xgs[2][16 * Npad];  // TDM double buffer
#endif

  int dir = blockIdx.x & 1, btile = blockIdx.x >> 1;
  int bbase = btile * 16;
  int tid = threadIdx.x, lane = tid & 31, wave = tid >> 5;

  for (int i = tid; i < 16 * 64; i += 416) {
    int m = i >> 6, j = i & 63;
    float hv = (j < Hsz) ? h0[((long)dir * Bsz + bbase + m) * Hsz + j] : 0.0f;
    hA[i] = (_Float16)hv;
    if (j < Hsz) cbuf[m * 52 + j] = c0[((long)dir * Bsz + bbase + m) * Hsz + j];
  }
  __syncthreads();

  int n0 = wave * 16;
  // resident w_hh B fragments (K = 0..31, 32..63)
  v16h bw0 = load_b_frag(WHH + ((long)dir * Npad + n0) * 64 + 0, 64);
  v16h bw1 = load_b_frag(WHH + ((long)dir * Npad + n0) * 64 + 32, 64);
  int nloc = lane & 15, hi2 = lane >> 4;
  const _Float16* xg = XG + (long)dir * BT * Npad;
  const unsigned long xg_row_stride = (unsigned long)Tlen * Npad;  // elements

#ifdef HAVE_TDM
  if (tid < 32) {  // prologue: prefetch tile for s=0
    int t0 = dir ? (Tlen - 1) : 0;
    tdm_load_2d(lds_off_of(&xgs[0][0]),
                xg + ((long)bbase * Tlen + t0) * Npad,
                (unsigned)Npad, 16u, xg_row_stride);
  }
#endif

  for (int s = 0; s < Tlen; ++s) {
    int t = dir ? (Tlen - 1 - s) : s;
#ifdef HAVE_TDM
    if (tid < 32) {
      if (s + 1 < Tlen) {  // issue next tile, retire the current one
        int tn = dir ? (Tlen - 2 - s) : (s + 1);
        tdm_load_2d(lds_off_of(&xgs[(s + 1) & 1][0]),
                    xg + ((long)bbase * Tlen + tn) * Npad,
                    (unsigned)Npad, 16u, xg_row_stride);
        __builtin_amdgcn_s_wait_tensorcnt(1);
      } else {
        __builtin_amdgcn_s_wait_tensorcnt(0);
      }
    }
    __syncthreads();  // xgs[s&1] now visible to all waves
#endif
    // hg = h_prev @ w_hh^T  (K padded 50->64 -> 2 WMMAs)
    v16h a0 = load_a_frag(hA, 64);
    v16h a1 = load_a_frag(hA + 32, 64);
    v8f acc = {};
    acc = wmma_f16(a0, bw0, acc);
    acc = wmma_f16(a1, bw1, acc);
#pragma unroll
    for (int r = 0; r < 8; ++r) {
      int m = r + hi2 * 8;
#ifdef HAVE_TDM
      float xv = (float)xgs[s & 1][m * Npad + n0 + nloc];
#else
      float xv = (float)xg[(((long)(bbase + m)) * Tlen + t) * Npad + n0 + nloc];
#endif
      gbuf[m * Npad + n0 + nloc] = acc[r] + xv;  // xg already holds bias
    }
    __syncthreads();
    // gate nonlinearity + state update over 16x50 (i,f,g,o in blocks of 50)
    for (int i = tid; i < 16 * Hsz; i += 416) {
      int m = i / Hsz, j = i % Hsz;
      float gi = gbuf[m * Npad + j];
      float gf = gbuf[m * Npad + 50 + j];
      float gg = gbuf[m * Npad + 100 + j];
      float go = gbuf[m * Npad + 150 + j];
      float c = cbuf[m * 52 + j];
      c = sigmoidf(gf) * c + sigmoidf(gi) * tanhf(gg);
      float h = sigmoidf(go) * tanhf(c);
      cbuf[m * 52 + j] = c;
      hA[m * 64 + j] = (_Float16)h;
      HC[(((long)(bbase + m)) * Tlen + t) * 128 + dir * Hsz + j] = (_Float16)h;
    }
    if (dir == 0) {  // zero pad columns 100..127 of the concat buffer
      for (int i = tid; i < 16 * 28; i += 416) {
        int m = i / 28, cI = 100 + i % 28;
        HC[(((long)(bbase + m)) * Tlen + t) * 128 + cI] = (_Float16)0.0f;
      }
    }
    __syncthreads();
  }
}

// ---- 4) tag projection + Viterbi, one wave32 per batch row ----------------
__global__ void __launch_bounds__(32)
viterbi(const _Float16* __restrict__ HC, const float* __restrict__ wtag,
        const float* __restrict__ btag, const float* __restrict__ trans,
        const float* __restrict__ mask, signed char* __restrict__ BP,
        float* __restrict__ out) {
  __shared__ float wt[7 * 100];
  __shared__ float tr[49];
  __shared__ float bt[8];
  __shared__ float sc[8];
  __shared__ float hrow[104];
  int b = blockIdx.x, lane = threadIdx.x;

  for (int i = lane; i < 700; i += 32) wt[i] = wtag[i];
  for (int i = lane; i < 49; i += 32) tr[i] = trans[i];
  if (lane < 7) bt[lane] = btag[lane];
  if (lane < 8) sc[lane] = (lane == 5) ? 0.0f : NEGF;  // START=5
  __syncthreads();

  for (int t = 0; t < Tlen; ++t) {
    for (int i = lane; i < 100; i += 32)
      hrow[i] = (float)HC[((long)b * Tlen + t) * 128 + i];
    __syncthreads();
    float mk = mask[(long)b * Tlen + t];
    float feat = 0.0f;
    if (lane < 7) {
      float a = bt[lane];
      for (int e = 0; e < 100; ++e) a += wt[lane * 100 + e] * hrow[e];
      feat = a * mk;
    }
    float prev[7];
#pragma unroll
    for (int j = 0; j < 7; ++j) prev[j] = sc[j];
    __syncthreads();
    if (lane < 7) {
      float best = -3.4e38f; int bj = 0;
#pragma unroll
      for (int j = 0; j < 7; ++j) {
        float v = prev[j] + tr[lane * 7 + j];
        if (v > best) { best = v; bj = j; }
      }
      BP[((long)t * Bsz + b) * 8 + lane] = (signed char)bj;
      sc[lane] = best + feat;
    }
    __syncthreads();
  }

  if (lane == 0) {
    float best = sc[0]; int bi = 0;
    for (int i = 1; i < 7; ++i) if (sc[i] > best) { best = sc[i]; bi = i; }
    out[b] = best;
    int tag = bi;
    for (int t = Tlen - 1; t >= 0; --t) {
      out[Bsz + (long)b * Tlen + t] = (float)tag;
      tag = BP[((long)t * Bsz + b) * 8 + tag];
    }
  }
}

// ---------------------------------------------------------------------------
extern "C" void kernel_launch(void* const* d_in, const int* in_sizes, int n_in,
                              void* d_out, int out_size, void* d_ws,
                              size_t ws_size, hipStream_t stream) {
  const int*   sent = (const int*)d_in[0];
  const float* mask = (const float*)d_in[1];
  const float* emb  = (const float*)d_in[2];
  const float* h0   = (const float*)d_in[3];
  const float* c0   = (const float*)d_in[4];
  const float* wih[3] = {(const float*)d_in[5], (const float*)d_in[8],  (const float*)d_in[11]};
  const float* whh[3] = {(const float*)d_in[6], (const float*)d_in[9],  (const float*)d_in[12]};
  const float* bia[3] = {(const float*)d_in[7], (const float*)d_in[10], (const float*)d_in[13]};
  const float* wtag  = (const float*)d_in[14];
  const float* btag  = (const float*)d_in[15];
  const float* trans = (const float*)d_in[16];
  float* out = (float*)d_out;

  // workspace carve-out (~131 MB, 256B aligned blocks)
  char* ws = (char*)d_ws;
  size_t off = 0;
  auto carve = [&](size_t bytes) -> void* {
    void* p = ws + off;
    off = (off + bytes + 255) & ~(size_t)255;
    return p;
  };
  _Float16* XPAD = (_Float16*)carve(BT * 320 * 2);        // layer-0 input
  _Float16* HA   = (_Float16*)carve(BT * 128 * 2);        // hcat ping
  _Float16* HB   = (_Float16*)carve(BT * 128 * 2);        // hcat pong
  _Float16* XG   = (_Float16*)carve(2 * BT * Npad * 2);   // gate preacts f16
  _Float16* WIH  = (_Float16*)carve(2 * Npad * 320 * 2);  // packed w_ih
  _Float16* WHHp = (_Float16*)carve(2 * Npad * 64 * 2);   // packed w_hh
  float*    BIA  = (float*)carve(2 * Npad * 4);           // padded bias
  signed char* BP = (signed char*)carve((size_t)Tlen * Bsz * 8);  // backptrs

  emb_gather<<<(int)(BT / 8), 256, 0, stream>>>(sent, emb, XPAD);

  const _Float16* lin[3] = {XPAD, HA, HB};
  _Float16* lout[3] = {HA, HB, HA};
  const int kin[3]  = {Esz, 2 * Hsz, 2 * Hsz};
  const int kpad[3] = {320, 128, 128};
  for (int L = 0; L < 3; ++L) {
    pack_weights<<<64, 256, 0, stream>>>(wih[L], whh[L], bia[L],
                                         WIH, WHHp, BIA, kin[L], kpad[L]);
    size_t smem = (size_t)128 * kpad[L] * sizeof(_Float16);  // 80KB / 32KB LDS
    gemm_xg<<<(int)(BT / 128), 256, smem, stream>>>(lin[L], WIH, BIA, XG,
                                                    kpad[L]);
    lstm_layer<<<16, 416, 0, stream>>>(XG, WHHp,
                                       h0 + (size_t)2 * L * Bsz * Hsz,
                                       c0 + (size_t)2 * L * Bsz * Hsz,
                                       lout[L]);
  }
  viterbi<<<Bsz, 32, 0, stream>>>(lout[2], wtag, btag, trans, mask, BP, out);
}
